// LabelIndependent_20993800143244
// MI455X (gfx1250) — matrix operand order repaired and checked by reading, not compiled, not run on hardware
//
#include <hip/hip_runtime.h>

#define BB 64
#define SS 512
#define HH 768
#define CC 50
#define MMM 2000
#define TT (BB * SS)

typedef __attribute__((ext_vector_type(16))) __bf16 v16bf;
typedef __attribute__((ext_vector_type(8)))  __bf16 v8bf;
typedef __attribute__((ext_vector_type(8)))  float  v8f;

__device__ __forceinline__ float wave_sum(float v) {
#pragma unroll
  for (int off = 16; off > 0; off >>= 1) v += __shfl_xor(v, off, 32);
  return v;
}

// ---------------------------------------------------------------------------
// 1) Per-token scalar scores: nts/mts/uts[t] = dot(se[t], w_*) + b_*
// ---------------------------------------------------------------------------
__global__ void token_scores(const float* __restrict__ se,
                             const float* __restrict__ wn, const float* __restrict__ bn,
                             const float* __restrict__ wm, const float* __restrict__ bm,
                             const float* __restrict__ wu, const float* __restrict__ bu,
                             float* __restrict__ nts, float* __restrict__ mts,
                             float* __restrict__ uts) {
  int t = blockIdx.x * 8 + (threadIdx.x >> 5);
  int lane = threadIdx.x & 31;
  const float* row = se + (size_t)t * HH;
  float a = 0.f, b = 0.f, c = 0.f;
  for (int h = lane; h < HH; h += 32) {
    float x = row[h];
    a += x * wn[h]; b += x * wm[h]; c += x * wu[h];
  }
  a = wave_sum(a); b = wave_sum(b); c = wave_sum(c);
  if (lane == 0) { nts[t] = a + bn[0]; mts[t] = b + bm[0]; uts[t] = c + bu[0]; }
}

// ---------------------------------------------------------------------------
// 2) Online softmax row stats (max + sum) — one 256-thread block per row.
// ---------------------------------------------------------------------------
__global__ void row_stats(int mode, const float* __restrict__ mask,
                          const float* __restrict__ score,
                          float* __restrict__ rmax, float* __restrict__ rinv, int L) {
  int r = blockIdx.x, tid = threadIdx.x;
  const float* mrow;
  const float* srow;
  if (mode == 0) { int c = r >> 6, b = r & 63; mrow = mask + (size_t)c * L; srow = score + (size_t)b * L; }
  else           { mrow = mask + (size_t)r * L; srow = score; }
  float m = -__builtin_inff(), s = 0.f;
  for (int l = tid; l < L; l += 256) {
    float x = mrow[l] + srow[l];
    if (x > -__builtin_inff()) {
      if (x > m) { s *= __expf(m - x); m = x; }  // m==-inf -> s==0 stays 0
      s += __expf(x - m);
    }
  }
  __shared__ float sm[256], ss[256];
  sm[tid] = m; ss[tid] = s;
  __syncthreads();
  for (int st = 128; st > 0; st >>= 1) {
    if (tid < st) {
      float m1 = sm[tid], s1 = ss[tid], m2 = sm[tid + st], s2 = ss[tid + st];
      float mn = fmaxf(m1, m2), o = 0.f;
      if (s1 > 0.f) o += s1 * __expf(m1 - mn);
      if (s2 > 0.f) o += s2 * __expf(m2 - mn);
      sm[tid] = mn; ss[tid] = o;
    }
    __syncthreads();
  }
  if (tid == 0) { rmax[r] = sm[0]; rinv[r] = ss[0] > 0.f ? 1.f / ss[0] : 0.f; }
}

// ---------------------------------------------------------------------------
// 3) Materialize attention weights as bf16 (A operand of the WMMA GEMMs).
// ---------------------------------------------------------------------------
__global__ void conv_attn(int mode, const float* __restrict__ mask,
                          const float* __restrict__ score,
                          const float* __restrict__ rmax, const float* __restrict__ rinv,
                          __bf16* __restrict__ outA, int L, int validC) {
  int rr = blockIdx.y;
  int l = blockIdx.x * 256 + threadIdx.x;
  if (l >= L) return;
  float w;
  if (mode == 0) {
    int b = rr >> 6, c = rr & 63;
    if (c >= validC) w = 0.f;
    else {
      int sr = c * 64 + b;
      float ri = rinv[sr];
      w = (ri == 0.f) ? 0.f
          : __expf(mask[(size_t)c * L + l] + score[(size_t)b * L + l] - rmax[sr]) * ri;
    }
  } else {
    if (rr >= validC) w = 0.f;
    else {
      float ri = rinv[rr];
      w = (ri == 0.f) ? 0.f
          : __expf(mask[(size_t)rr * L + l] + score[l] - rmax[rr]) * ri;
    }
  }
  outA[(size_t)rr * L + l] = (__bf16)w;
}

// ---------------------------------------------------------------------------
// 4) story [T,H] f32 -> seT [H,T] bf16 via LDS tile transpose.
// ---------------------------------------------------------------------------
__global__ void transpose_bf16(const float* __restrict__ se, __bf16* __restrict__ seT) {
  __shared__ float tile[32][33];
  int h0 = blockIdx.x * 32, t0 = blockIdx.y * 32;
  int tx = threadIdx.x, ty = threadIdx.y;  // 32 x 8
#pragma unroll
  for (int i = 0; i < 32; i += 8)
    tile[ty + i][tx] = se[(size_t)(t0 + ty + i) * HH + h0 + tx];
  __syncthreads();
#pragma unroll
  for (int i = 0; i < 32; i += 8)
    seT[(size_t)(h0 + ty + i) * TT + t0 + tx] = (__bf16)tile[tx][ty + i];
}

// ---------------------------------------------------------------------------
// 5) WMMA GEMM: out[r, n] = sum_k A[r,k] * seT[n,k]   (both bf16, f32 acc)
//    One wave computes a 32x64 tile: 2 A-frags x 4 B-frags -> 8 WMMAs / k-step.
//    Software-pipelined: next k-step fragments are loaded before the current
//    WMMAs are issued so waits are partial, not a drain to loadcnt==0.
//    B operand is always seT with row stride TT (compile-time), so the four
//    B rows per frag group become immediate offsets (no per-row addr adds).
// ---------------------------------------------------------------------------
struct Frags {
  v16bf a0, a1, b0, b1, b2, b3;
};

__device__ __forceinline__ v16bf load_afrag(const __bf16* p, int kb, int khalf) {
  v8bf lo = *(const v8bf*)(p + kb + khalf * 16);
  v8bf hi = *(const v8bf*)(p + kb + khalf * 16 + 8);
  return __builtin_shufflevector(lo, hi, 0, 1, 2, 3, 4, 5, 6, 7,
                                 8, 9, 10, 11, 12, 13, 14, 15);
}

__global__ void attn_wmma_gemm(const __bf16* __restrict__ A, int lda, int aRowsPerZ,
                               const __bf16* __restrict__ Bm, int kColOffPerZ,
                               float* __restrict__ out, int ldo, int outPerZ,
                               int Rstore, int K) {
  const int lane = threadIdx.x;
  const int khalf = lane >> 4;
  const int l16 = lane & 15;
  const int mt = blockIdx.x;  // 32-row tile
  const int nt = blockIdx.y;  // 64-col tile
  const int z = blockIdx.z;

  const __bf16* Arow0 = A + (size_t)(z * aRowsPerZ + mt * 32 + l16) * lda;
  const __bf16* Arow1 = Arow0 + (size_t)16 * lda;
  // Base B row for this lane; rows 16/32/48 are reached via constant offsets.
  const __bf16* Brow = Bm + (size_t)(nt * 64 + l16) * TT + (size_t)z * kColOffPerZ;

  auto load_frags = [&](int kb) -> Frags {
    Frags f;
    f.a0 = load_afrag(Arow0, kb, khalf);
    f.a1 = load_afrag(Arow1, kb, khalf);
    const __bf16* bk = Brow + kb + khalf * 16;
    f.b0 = *(const v16bf*)(bk);
    f.b1 = *(const v16bf*)(bk + (size_t)16 * TT);
    f.b2 = *(const v16bf*)(bk + (size_t)32 * TT);
    f.b3 = *(const v16bf*)(bk + (size_t)48 * TT);
    return f;
  };

  v8f acc00 = {}, acc01 = {}, acc02 = {}, acc03 = {};
  v8f acc10 = {}, acc11 = {}, acc12 = {}, acc13 = {};

  Frags cur = load_frags(0);
#pragma unroll 2
  for (int kb = 0; kb + 32 < K; kb += 32) {
    __builtin_prefetch(Arow0 + kb + 288, 0, 1);
    __builtin_prefetch(Arow1 + kb + 288, 0, 1);
    Frags nxt = load_frags(kb + 32);
    acc00 = __builtin_amdgcn_wmma_f32_16x16x32_bf16(false, cur.a0, false, cur.b0, (short)0, acc00, false, false);
    acc01 = __builtin_amdgcn_wmma_f32_16x16x32_bf16(false, cur.a0, false, cur.b1, (short)0, acc01, false, false);
    acc02 = __builtin_amdgcn_wmma_f32_16x16x32_bf16(false, cur.a0, false, cur.b2, (short)0, acc02, false, false);
    acc03 = __builtin_amdgcn_wmma_f32_16x16x32_bf16(false, cur.a0, false, cur.b3, (short)0, acc03, false, false);
    acc10 = __builtin_amdgcn_wmma_f32_16x16x32_bf16(false, cur.a1, false, cur.b0, (short)0, acc10, false, false);
    acc11 = __builtin_amdgcn_wmma_f32_16x16x32_bf16(false, cur.a1, false, cur.b1, (short)0, acc11, false, false);
    acc12 = __builtin_amdgcn_wmma_f32_16x16x32_bf16(false, cur.a1, false, cur.b2, (short)0, acc12, false, false);
    acc13 = __builtin_amdgcn_wmma_f32_16x16x32_bf16(false, cur.a1, false, cur.b3, (short)0, acc13, false, false);
    cur = nxt;
  }
  acc00 = __builtin_amdgcn_wmma_f32_16x16x32_bf16(false, cur.a0, false, cur.b0, (short)0, acc00, false, false);
  acc01 = __builtin_amdgcn_wmma_f32_16x16x32_bf16(false, cur.a0, false, cur.b1, (short)0, acc01, false, false);
  acc02 = __builtin_amdgcn_wmma_f32_16x16x32_bf16(false, cur.a0, false, cur.b2, (short)0, acc02, false, false);
  acc03 = __builtin_amdgcn_wmma_f32_16x16x32_bf16(false, cur.a0, false, cur.b3, (short)0, acc03, false, false);
  acc10 = __builtin_amdgcn_wmma_f32_16x16x32_bf16(false, cur.a1, false, cur.b0, (short)0, acc10, false, false);
  acc11 = __builtin_amdgcn_wmma_f32_16x16x32_bf16(false, cur.a1, false, cur.b1, (short)0, acc11, false, false);
  acc12 = __builtin_amdgcn_wmma_f32_16x16x32_bf16(false, cur.a1, false, cur.b2, (short)0, acc12, false, false);
  acc13 = __builtin_amdgcn_wmma_f32_16x16x32_bf16(false, cur.a1, false, cur.b3, (short)0, acc13, false, false);

  float* obase = out + (size_t)z * outPerZ;
#pragma unroll
  for (int r = 0; r < 8; ++r) {
    int row0 = mt * 32 + khalf * 8 + r;
    if (row0 < Rstore) {
      float* orow = obase + (size_t)row0 * ldo + nt * 64 + l16;
      orow[0] = acc00[r]; orow[16] = acc01[r]; orow[32] = acc02[r]; orow[48] = acc03[r];
    }
    int row1 = row0 + 16;
    if (row1 < Rstore) {
      float* orow = obase + (size_t)row1 * ldo + nt * 64 + l16;
      orow[0] = acc10[r]; orow[16] = acc11[r]; orow[32] = acc12[r]; orow[48] = acc13[r];
    }
  }
}

// ---------------------------------------------------------------------------
// 6) name_e: softmax over blocks b of dot(names_emb[b,c], w_name), weighted sum
// ---------------------------------------------------------------------------
__global__ void name_combine(const float* __restrict__ names_emb,  // [B][64][H]
                             const float* __restrict__ w_name, const float* __restrict__ b_name,
                             float* __restrict__ name_e) {         // [64][H]
  __shared__ float sc[BB];
  __shared__ float p[BB];
  int c = blockIdx.x, tid = threadIdx.x;
  int lane = tid & 31, w = tid >> 5;
  for (int b = w; b < BB; b += 8) {
    const float* e = names_emb + ((size_t)b * 64 + c) * HH;
    float s = 0.f;
    for (int h = lane; h < HH; h += 32) s += e[h] * w_name[h];
    s = wave_sum(s);
    if (lane == 0) sc[b] = s + b_name[0];
  }
  __syncthreads();
  if (tid == 0) {
    float mx = -__builtin_inff();
    for (int b = 0; b < BB; ++b) mx = fmaxf(mx, sc[b]);
    float den = 0.f;
    for (int b = 0; b < BB; ++b) { p[b] = __expf(sc[b] - mx); den += p[b]; }
    float inv = 1.f / den;
    for (int b = 0; b < BB; ++b) p[b] *= inv;
  }
  __syncthreads();
  for (int h = tid; h < HH; h += 256) {
    float acc = 0.f;
    for (int b = 0; b < BB; ++b) acc += p[b] * names_emb[((size_t)b * 64 + c) * HH + h];
    name_e[(size_t)c * HH + h] = acc;
  }
}

// 7a) mention scores (wave per mention)
__global__ void mention_score(const float* __restrict__ m_emb, const float* __restrict__ w_men,
                              const float* __restrict__ b_men, float* __restrict__ m_score) {
  int m = blockIdx.x * 8 + (threadIdx.x >> 5);
  int lane = threadIdx.x & 31;
  if (m >= MMM) return;
  const float* e = m_emb + (size_t)m * HH;
  float s = 0.f;
  for (int h = lane; h < HH; h += 32) s += e[h] * w_men[h];
  s = wave_sum(s);
  if (lane == 0) m_score[m] = s + b_men[0];
}

// 7b) segment max/den/count per character
__global__ void seg_stats(const float* __restrict__ m_score, const int* __restrict__ seg,
                          float* __restrict__ segMax, float* __restrict__ segDen,
                          float* __restrict__ segCnt) {
  int c = threadIdx.x;
  float mx = -__builtin_inff();
  int cnt = 0;
  for (int m = 0; m < MMM; ++m)
    if (seg[m] == c) { mx = fmaxf(mx, m_score[m]); ++cnt; }
  float den = 0.f;
  if (cnt > 0)
    for (int m = 0; m < MMM; ++m)
      if (seg[m] == c) den += __expf(m_score[m] - mx);
  segMax[c] = mx; segDen[c] = den; segCnt[c] = (float)cnt;
}

__global__ void zero_f32(float* __restrict__ p, int n) {
  int i = blockIdx.x * 256 + threadIdx.x;
  if (i < n) p[i] = 0.f;
}

// 7c) mention_e[c] += mw[m] * m_emb[m]  (atomic scatter, block per mention)
__global__ void mention_scatter(const float* __restrict__ m_emb, const float* __restrict__ m_score,
                                const int* __restrict__ seg, const float* __restrict__ segMax,
                                const float* __restrict__ segDen, float* __restrict__ mention_e) {
  int m = blockIdx.x;
  int c = seg[m];
  float den = segDen[c];
  float w = den > 0.f ? __expf(m_score[m] - segMax[c]) / den : 0.f;
  for (int h = threadIdx.x; h < HH; h += 256)
    atomicAdd(&mention_e[(size_t)c * HH + h], w * m_emb[(size_t)m * HH + h]);
}

// 8) final gated 3-way softmax combine
__global__ void final_combine(const float* __restrict__ name_e, const float* __restrict__ mention_e,
                              const float* __restrict__ utt_e, const float* __restrict__ segCnt,
                              const float* __restrict__ uttInv, const float* __restrict__ w_comb,
                              const float* __restrict__ b_comb, float* __restrict__ out) {
  int c = blockIdx.x, tid = threadIdx.x;
  const float* e0 = name_e + (size_t)c * HH;
  const float* e1 = mention_e + (size_t)c * HH;
  const float* e2 = utt_e + (size_t)c * HH;
  float s0 = 0.f, s1 = 0.f, s2 = 0.f;
  for (int h = tid; h < HH; h += 256) {
    float w = w_comb[h];
    s0 += e0[h] * w; s1 += e1[h] * w; s2 += e2[h] * w;
  }
  __shared__ float r0[256], r1[256], r2[256];
  r0[tid] = s0; r1[tid] = s1; r2[tid] = s2;
  __syncthreads();
  for (int st = 128; st > 0; st >>= 1) {
    if (tid < st) { r0[tid] += r0[tid + st]; r1[tid] += r1[tid + st]; r2[tid] += r2[tid + st]; }
    __syncthreads();
  }
  __shared__ float p[3];
  if (tid == 0) {
    float bb = b_comb[0];
    float a0 = r0[0] + bb;
    float a1 = (segCnt[c] > 0.f) ? r1[0] + bb : -__builtin_inff();
    float a2 = (uttInv[c] > 0.f) ? r2[0] + bb : -__builtin_inff();
    float mx = fmaxf(a0, fmaxf(a1, a2));
    float x0 = __expf(a0 - mx);
    float x1 = (a1 == -__builtin_inff()) ? 0.f : __expf(a1 - mx);
    float x2 = (a2 == -__builtin_inff()) ? 0.f : __expf(a2 - mx);
    float inv = 1.f / (x0 + x1 + x2);
    p[0] = x0 * inv; p[1] = x1 * inv; p[2] = x2 * inv;
  }
  __syncthreads();
  for (int h = tid; h < HH; h += 256)
    out[(size_t)c * HH + h] = p[0] * e0[h] + p[1] * e1[h] + p[2] * e2[h];
}

// ---------------------------------------------------------------------------
extern "C" void kernel_launch(void* const* d_in, const int* in_sizes, int n_in,
                              void* d_out, int out_size, void* d_ws, size_t ws_size,
                              hipStream_t stream) {
  const float* se = (const float*)d_in[0];
  const float* namesM = (const float*)d_in[1];
  const float* uttM = (const float*)d_in[2];
  const float* menM = (const float*)d_in[3];
  const int* seg = (const int*)d_in[4];
  const float* w_name_tok = (const float*)d_in[5];
  const float* b_name_tok = (const float*)d_in[6];
  const float* w_name = (const float*)d_in[7];
  const float* b_name = (const float*)d_in[8];
  const float* w_men_tok = (const float*)d_in[9];
  const float* b_men_tok = (const float*)d_in[10];
  const float* w_men = (const float*)d_in[11];
  const float* b_men = (const float*)d_in[12];
  const float* w_utt = (const float*)d_in[13];
  const float* b_utt = (const float*)d_in[14];
  const float* w_comb = (const float*)d_in[15];
  const float* b_comb = (const float*)d_in[16];

  char* ws = (char*)d_ws;
  size_t cur = 0;
  auto alloc = [&](size_t bytes) -> void* {
    void* p = ws + cur;
    cur = (cur + bytes + 255) & ~(size_t)255;
    return p;
  };

  float* nts = (float*)alloc((size_t)TT * 4);
  float* mts = (float*)alloc((size_t)TT * 4);
  float* uts = (float*)alloc((size_t)TT * 4);
  float* nameMax = (float*)alloc((size_t)CC * BB * 4);
  float* nameInv = (float*)alloc((size_t)CC * BB * 4);
  float* menMax = (float*)alloc((size_t)MMM * 4);
  float* menInv = (float*)alloc((size_t)MMM * 4);
  float* uttMax = (float*)alloc(64 * 4);
  float* uttInv = (float*)alloc(64 * 4);
  __bf16* n_attnb = (__bf16*)alloc((size_t)BB * 64 * SS * 2);
  __bf16* m_attnb = (__bf16*)alloc((size_t)MMM * TT * 2);
  __bf16* u_attnb = (__bf16*)alloc((size_t)64 * TT * 2);
  __bf16* seT = (__bf16*)alloc((size_t)HH * TT * 2);
  float* names_emb = (float*)alloc((size_t)BB * 64 * HH * 4);
  float* m_emb = (float*)alloc((size_t)MMM * HH * 4);
  float* utt_e = (float*)alloc((size_t)64 * HH * 4);
  float* m_score = (float*)alloc((size_t)MMM * 4);
  float* segMax = (float*)alloc(64 * 4);
  float* segDen = (float*)alloc(64 * 4);
  float* segCnt = (float*)alloc(64 * 4);
  float* name_e = (float*)alloc((size_t)64 * HH * 4);
  float* mention_e = (float*)alloc((size_t)64 * HH * 4);

  // 1) token scores
  token_scores<<<TT / 8, 256, 0, stream>>>(se, w_name_tok, b_name_tok, w_men_tok,
                                           b_men_tok, w_utt, b_utt, nts, mts, uts);
  // 2) softmax row stats
  row_stats<<<CC * BB, 256, 0, stream>>>(0, namesM, nts, nameMax, nameInv, SS);
  row_stats<<<MMM, 256, 0, stream>>>(1, menM, mts, menMax, menInv, TT);
  row_stats<<<CC, 256, 0, stream>>>(2, uttM, uts, uttMax, uttInv, TT);
  // 3) attention weights -> bf16
  conv_attn<<<dim3(SS / 256, BB * 64), 256, 0, stream>>>(0, namesM, nts, nameMax, nameInv,
                                                         n_attnb, SS, CC);
  conv_attn<<<dim3(TT / 256, MMM), 256, 0, stream>>>(1, menM, mts, menMax, menInv,
                                                     m_attnb, TT, MMM);
  conv_attn<<<dim3(TT / 256, 64), 256, 0, stream>>>(2, uttM, uts, uttMax, uttInv,
                                                    u_attnb, TT, CC);
  // 4) story -> bf16 transposed [H, T]
  transpose_bf16<<<dim3(HH / 32, TT / 32), dim3(32, 8), 0, stream>>>(se, seT);
  // 5) the three attention-weighted GEMMs on WMMA (32x64 tiles per wave)
  //    names: per-block [64(C pad),S] x [S,H] -> names_emb[b][64][H]
  attn_wmma_gemm<<<dim3(2, HH / 64, BB), 32, 0, stream>>>(n_attnb, SS, 64, seT, SS,
                                                          names_emb, HH, 64 * HH, CC, SS);
  //    mentions: [2000, T] x [T, H] -> m_emb  (dominant: ~100 GFLOP)
  attn_wmma_gemm<<<dim3((MMM + 31) / 32, HH / 64, 1), 32, 0, stream>>>(m_attnb, TT, 0, seT, 0,
                                                                       m_emb, HH, 0, MMM, TT);
  //    utterances: [64(C pad), T] x [T, H] -> utt_e
  attn_wmma_gemm<<<dim3(2, HH / 64, 1), 32, 0, stream>>>(u_attnb, TT, 0, seT, 0,
                                                         utt_e, HH, 0, CC, TT);
  // 6) name representation
  name_combine<<<CC, 256, 0, stream>>>(names_emb, w_name, b_name, name_e);
  // 7) mention representation (segment softmax)
  mention_score<<<MMM / 8, 256, 0, stream>>>(m_emb, w_men, b_men, m_score);
  seg_stats<<<1, 64, 0, stream>>>(m_score, seg, segMax, segDen, segCnt);
  zero_f32<<<(64 * HH + 255) / 256, 256, 0, stream>>>(mention_e, 64 * HH);
  mention_scatter<<<MMM, 256, 0, stream>>>(m_emb, m_score, seg, segMax, segDen, mention_e);
  // 8) final combine -> d_out [C, H]
  final_combine<<<CC, 256, 0, stream>>>(name_e, mention_e, utt_e, segCnt, uttInv,
                                        w_comb, b_comb, (float*)d_out);
}
